// FirstCollectionLayer_28905129902362
// MI455X (gfx1250) — matrix operand order repaired and checked by reading, not compiled, not run on hardware
//
#include <hip/hip_runtime.h>
#include <hip/hip_bf16.h>
#include <math.h>

typedef __attribute__((ext_vector_type(16))) __bf16 v16bf;
typedef __attribute__((ext_vector_type(8)))  __bf16 v8bf;
typedef __attribute__((ext_vector_type(4)))  __bf16 v4bf;
typedef __attribute__((ext_vector_type(8)))  float  v8f;

#define LDS_PAD 40   // bf16 elems per LDS row: 80B row stride (16B aligned, bank stride 20)

// ---------------------------------------------------------------------------
// f32 -> bf16 convert
// ---------------------------------------------------------------------------
__global__ __launch_bounds__(256)
void cvt_bf16_kernel(const float* __restrict__ x, __bf16* __restrict__ y, long n) {
    long i = (long)blockIdx.x * blockDim.x + threadIdx.x;
    if (i < n) y[i] = (__bf16)x[i];
}

// ---------------------------------------------------------------------------
// Row LayerNorm: one block per row. Output type selected at compile time.
// ---------------------------------------------------------------------------
template <bool TO_BF16>
__global__ __launch_bounds__(256)
void ln_kernel(const float* __restrict__ x, const float* __restrict__ g,
               const float* __restrict__ bta, void* __restrict__ outp, int D) {
    long row = blockIdx.x;
    const float* xr = x + row * (long)D;
    __shared__ float red[256];
    int tid = threadIdx.x;
    float s = 0.f;
    for (int i = tid; i < D; i += 256) s += xr[i];
    red[tid] = s; __syncthreads();
    for (int o = 128; o > 0; o >>= 1) { if (tid < o) red[tid] += red[tid + o]; __syncthreads(); }
    float mu = red[0] / (float)D;
    __syncthreads();
    float v = 0.f;
    for (int i = tid; i < D; i += 256) { float d = xr[i] - mu; v += d * d; }
    red[tid] = v; __syncthreads();
    for (int o = 128; o > 0; o >>= 1) { if (tid < o) red[tid] += red[tid + o]; __syncthreads(); }
    float inv = rsqrtf(red[0] / (float)D + 1e-5f);
    for (int i = tid; i < D; i += 256) {
        float y = (xr[i] - mu) * inv * g[i] + bta[i];
        if (TO_BF16) ((__bf16*)outp)[row * (long)D + i] = (__bf16)y;
        else         ((float*)outp)[row * (long)D + i] = y;
    }
}

// ---------------------------------------------------------------------------
// Generic bf16 GEMM: C[M,N] (+)= A[M,Kd] @ B[Kd,N] + bias[N]
// Tile 64x64, BK=32, 8 waves; v_wmma_f32_16x16x32_bf16.
// Fragment layouts per CDNA5 ISA 7.12.2:
//   A  (16x32 bf16): lane holds row M=lane%16; elem i -> K = i + 8*(i>>3) + 8*(lane>>4)
//   B  (32x16 bf16): lane holds col N=lane%16; elem i -> K = i + 8*(i>>3) + 8*(lane>>4)
//   CD (16x16 f32) : elem r -> (M = r + 8*(lane>>4), N = lane%16)
// B tile stored K-innermost (transposed) in LDS so both fragments read
// K-contiguous 16B chunks (ds_load_b128). Compile-time epilogue flags.
// ---------------------------------------------------------------------------
template <bool HAS_BIAS, bool HAS_CF, bool HAS_CBF, bool ACCUM>
__global__ __launch_bounds__(256)
void gemm_bf16_kernel(const __bf16* __restrict__ A, const __bf16* __restrict__ Bm,
                      const float* __restrict__ bias,
                      float* __restrict__ Cf, __bf16* __restrict__ Cbf,
                      int N, int Kd, long sA, long sB, long sC) {
    __shared__ __bf16 sa[64][LDS_PAD];
    __shared__ __bf16 sb[64][LDS_PAD];   // [n][k]
    int b = blockIdx.z;
    const __bf16* Ab = A + (long)b * sA;
    const __bf16* Bb = Bm + (long)b * sB;
    long cb = (long)b * sC;
    int m0 = blockIdx.y * 64;
    int n0 = blockIdx.x * 64;
    int tid = threadIdx.x;
    int wave = tid >> 5, lane = tid & 31;
    int hf = lane >> 4, ln16 = lane & 15;
    int wm = (wave >> 1) * 16;   // 0,16,32,48
    int wn = (wave & 1) * 32;    // 0,32

    const int ra = (tid * 8) >> 5, ca = (tid * 8) & 31;   // A stage coords
    const int rb = (tid * 8) >> 6, cbn = (tid * 8) & 63;  // B stage coords

    v8f acc[2];
#pragma unroll
    for (int nt = 0; nt < 2; ++nt)
#pragma unroll
        for (int r = 0; r < 8; ++r) acc[nt][r] = 0.f;

    for (int k0 = 0; k0 < Kd; k0 += 32) {
        const __bf16* ap = &Ab[(long)(m0 + ra) * Kd + k0 + ca];
        *(v8bf*)&sa[ra][ca] = *(const v8bf*)ap;          // one b128 load+store
        __builtin_prefetch(ap + 32, 0, 0);               // next K tile (speculative)
        const __bf16* bp = &Bb[(long)(k0 + rb) * N + n0 + cbn];
        v8bf bv = *(const v8bf*)bp;                      // one b128 load
        __builtin_prefetch(bp + (long)32 * N, 0, 0);
#pragma unroll
        for (int i = 0; i < 8; ++i) sb[cbn + i][rb] = bv[i];  // transpose scatter
        __syncthreads();

        v16bf af;
#pragma unroll
        for (int i = 0; i < 16; ++i) {
            int kk = i + 8 * (i >> 3) + 8 * hf;
            af[i] = sa[wm + ln16][kk];
        }
#pragma unroll
        for (int nt = 0; nt < 2; ++nt) {
            v16bf bf;
#pragma unroll
            for (int i = 0; i < 16; ++i) {
                int kk = i + 8 * (i >> 3) + 8 * hf;
                bf[i] = sb[wn + nt * 16 + ln16][kk];
            }
            acc[nt] = __builtin_amdgcn_wmma_f32_16x16x32_bf16(
                false, af, false, bf, (short)0, acc[nt], false, false);
        }
        __syncthreads();
    }

#pragma unroll
    for (int nt = 0; nt < 2; ++nt) {
        int col = n0 + wn + nt * 16 + ln16;
        float bv = HAS_BIAS ? bias[col] : 0.f;
#pragma unroll
        for (int r = 0; r < 8; ++r) {
            int row = m0 + wm + r + 8 * hf;
            long idx = cb + (long)row * N + col;
            float v = acc[nt][r] + bv;
            if (ACCUM)   v += Cf[idx];
            if (HAS_CF)  Cf[idx] = v;
            if (HAS_CBF) Cbf[idx] = (__bf16)v;
        }
    }
}

// ---------------------------------------------------------------------------
// Fused edge-score kernel, one block per (b, k, 32-row s tile).
// h = gelu(AS[s,:] + BT[k,:] + s_bf[s,:] @ (t_k (.) W1c));
// e = softsign(h . m2_w + m2_b) * mask.  Writes e (f32, [B,S,K]) and
// e^T (bf16, [B,K,S]) for the downstream einsum GEMMs.
// Wk slice is materialized in LDS per K-step: swk[j][e] = t_k[e]*W1c[e][j].
// ---------------------------------------------------------------------------
__global__ __launch_bounds__(256)
void edge_kernel(const __bf16* __restrict__ s_bf,  // [B*S, E]
                 const float* __restrict__ AS,     // [B*S, E]
                 const float* __restrict__ BT,     // [B*K, E]
                 const float* __restrict__ tvals,  // [B*K, E]
                 const float* __restrict__ w1c,    // [E, E]
                 const float* __restrict__ m2w,    // [E]
                 const float* __restrict__ m2b,    // [1]
                 const float* __restrict__ mask,   // [B, S]
                 float* __restrict__ eout,         // [B, S, K]
                 __bf16* __restrict__ ebT,         // [B, K, S]
                 int S, int K) {
    const int E = 256;
    __shared__ __bf16 sa[32][LDS_PAD];
    __shared__ __bf16 swk[256][LDS_PAD];   // [j][e]
    __shared__ float st[32];
    __shared__ float part[16][8];
    int b = blockIdx.z, k = blockIdx.y;
    int s0 = blockIdx.x * 32;
    int tid = threadIdx.x, wave = tid >> 5, lane = tid & 31;
    int hf = lane >> 4, ln16 = lane & 15;
    int wm = (wave >> 2) * 16;   // 0,16
    int wn = (wave & 3) * 64;    // 0,64,128,192
    long sbase = ((long)b * S + s0) * E;
    long tbase = ((long)b * K + k) * E;
    const int ra = (tid * 4) >> 5, ca = (tid * 4) & 31;

    v8f acc[4];
#pragma unroll
    for (int nt = 0; nt < 4; ++nt)
#pragma unroll
        for (int r = 0; r < 8; ++r) acc[nt][r] = 0.f;

    for (int e0 = 0; e0 < E; e0 += 32) {
        if (tid < 32) st[tid] = tvals[tbase + e0 + tid];
        __syncthreads();   // prior WMMA done; st visible
        {                  // s tile 32x32, one b64 load per thread
            const __bf16* sp = &s_bf[sbase + (long)ra * E + e0 + ca];
            *(v4bf*)&sa[ra][ca] = *(const v4bf*)sp;
            __builtin_prefetch(sp + 32, 0, 0);
        }
        {                  // Wk slice: column j = tid
            int j = tid;
#pragma unroll
            for (int e = 0; e < 32; ++e)
                swk[j][e] = (__bf16)(st[e] * w1c[(long)(e0 + e) * E + j]);
        }
        __syncthreads();

        v16bf af;
#pragma unroll
        for (int i = 0; i < 16; ++i) {
            int kk = i + 8 * (i >> 3) + 8 * hf;
            af[i] = sa[wm + ln16][kk];
        }
#pragma unroll
        for (int nt = 0; nt < 4; ++nt) {
            v16bf bf;
#pragma unroll
            for (int i = 0; i < 16; ++i) {
                int kk = i + 8 * (i >> 3) + 8 * hf;
                bf[i] = swk[wn + nt * 16 + ln16][kk];
            }
            acc[nt] = __builtin_amdgcn_wmma_f32_16x16x32_bf16(
                false, af, false, bf, (short)0, acc[nt], false, false);
        }
        __syncthreads();
    }

    // epilogue: gelu + dot with m2w, deterministic reduction over j
    float rowsum[8];
#pragma unroll
    for (int r = 0; r < 8; ++r) rowsum[r] = 0.f;
#pragma unroll
    for (int nt = 0; nt < 4; ++nt) {
        int j = wn + nt * 16 + ln16;
        float btv = BT[tbase + j];
        float m2v = m2w[j];
#pragma unroll
        for (int r = 0; r < 8; ++r) {
            int rl = wm + r + 8 * hf;
            float pre = acc[nt][r] + AS[sbase + (long)rl * E + j] + btv;
            float gl = 0.5f * pre * (1.f + erff(pre * 0.70710678118654752f));
            rowsum[r] += gl * m2v;
        }
    }
#pragma unroll
    for (int m = 1; m < 16; m <<= 1)
#pragma unroll
        for (int r = 0; r < 8; ++r) rowsum[r] += __shfl_xor(rowsum[r], m, 16);
    if (ln16 == 0)
#pragma unroll
        for (int r = 0; r < 8; ++r) part[wave * 2 + hf][r] = rowsum[r];
    __syncthreads();
    if (tid < 32) {
        int g4 = tid >> 4, hh = (tid >> 3) & 1, r = tid & 7;
        float e = 0.f;
#pragma unroll
        for (int w = 0; w < 4; ++w) e += part[(g4 * 4 + w) * 2 + hh][r];
        e += m2b[0];
        e = e / (1.f + fabsf(e));
        int s = s0 + tid;
        e *= mask[(long)b * S + s];
        eout[((long)b * S + s) * K + k] = e;
        ebT[((long)b * K + k) * S + s] = (__bf16)e;
    }
}

// ---------------------------------------------------------------------------
extern "C" void kernel_launch(void* const* d_in, const int* in_sizes, int n_in,
                              void* d_out, int out_size, void* d_ws, size_t ws_size,
                              hipStream_t stream) {
    (void)in_sizes; (void)n_in; (void)out_size; (void)ws_size;
    const int B = 2, St = 2048, Si = 1024, K = 64, D = 768, E = 256;
    const int NT = B * St, NI = B * Si, NK = B * K;

    const float* text   = (const float*)d_in[0];
    const float* image  = (const float*)d_in[1];
    const float* lstat0 = (const float*)d_in[2];
    const float* lval0  = (const float*)d_in[3];
    const float* tmask  = (const float*)d_in[4];
    const float* imask  = (const float*)d_in[5];
    const float *t_sn_g = (const float*)d_in[6],  *t_sn_b = (const float*)d_in[7];
    const float *t_tn_g = (const float*)d_in[8],  *t_tn_b = (const float*)d_in[9];
    const float *t_su_w = (const float*)d_in[10], *t_su_b = (const float*)d_in[11];
    const float *t_tp_w = (const float*)d_in[12], *t_tp_b = (const float*)d_in[13];
    const float *t_m1_w = (const float*)d_in[14], *t_m1_b = (const float*)d_in[15];
    const float *t_m2_w = (const float*)d_in[16], *t_m2_b = (const float*)d_in[17];
    const float *i_sn_g = (const float*)d_in[18], *i_sn_b = (const float*)d_in[19];
    const float *i_tn_g = (const float*)d_in[20], *i_tn_b = (const float*)d_in[21];
    const float *i_su_w = (const float*)d_in[22], *i_su_b = (const float*)d_in[23];
    const float *i_tp_w = (const float*)d_in[24], *i_tp_b = (const float*)d_in[25];
    const float *i_m1_w = (const float*)d_in[26], *i_m1_b = (const float*)d_in[27];
    const float *i_m2_w = (const float*)d_in[28], *i_m2_b = (const float*)d_in[29];
    const float *tsn_g = (const float*)d_in[30], *tsn_b = (const float*)d_in[31];
    const float *isn_g = (const float*)d_in[32], *isn_b = (const float*)d_in[33];
    const float *stn_g = (const float*)d_in[34], *stn_b = (const float*)d_in[35];
    const float *vn_g  = (const float*)d_in[36], *vn_b  = (const float*)d_in[37];
    const float *tsp_w = (const float*)d_in[38], *tsp_b = (const float*)d_in[39];
    const float *isp_w = (const float*)d_in[40], *isp_b = (const float*)d_in[41];

    float* out_state = (float*)d_out;                      // [B,K,D]
    float* out_value = out_state + (size_t)B * K * D;      // [B,K,D]
    float* out_t2l   = out_value + (size_t)B * K * D;      // [B,St,K]
    float* out_i2l   = out_t2l + (size_t)B * St * K;       // [B,Si,K]

    char* wsp = (char*)d_ws;
    size_t off = 0;
    auto alloc = [&](size_t bytes) -> void* {
        void* p = wsp + off;
        off += (bytes + 255) & ~(size_t)255;
        return p;
    };
    __bf16* xt_sn  = (__bf16*)alloc((size_t)NT * D * 2);
    __bf16* xt_tsn = (__bf16*)alloc((size_t)NT * D * 2);
    __bf16* xi_sn  = (__bf16*)alloc((size_t)NI * D * 2);
    __bf16* xi_isn = (__bf16*)alloc((size_t)NI * D * 2);
    __bf16* lt_t   = (__bf16*)alloc((size_t)NK * D * 2);
    __bf16* lt_i   = (__bf16*)alloc((size_t)NK * D * 2);
    __bf16* wsu_t  = (__bf16*)alloc((size_t)D * E * 2);
    __bf16* wtp_t  = (__bf16*)alloc((size_t)D * E * 2);
    __bf16* wsu_i  = (__bf16*)alloc((size_t)D * E * 2);
    __bf16* wtp_i  = (__bf16*)alloc((size_t)D * E * 2);
    __bf16* m1t_bf = (__bf16*)alloc((size_t)3 * E * E * 2);
    __bf16* m1i_bf = (__bf16*)alloc((size_t)3 * E * E * 2);
    __bf16* wtsp   = (__bf16*)alloc((size_t)D * D * 2);
    __bf16* wisp   = (__bf16*)alloc((size_t)D * D * 2);
    __bf16* s_t    = (__bf16*)alloc((size_t)NT * E * 2);
    __bf16* s_i    = (__bf16*)alloc((size_t)NI * E * 2);
    __bf16* tt_bf  = (__bf16*)alloc((size_t)NK * E * 2);
    __bf16* ti_bf  = (__bf16*)alloc((size_t)NK * E * 2);
    __bf16* tsf    = (__bf16*)alloc((size_t)NT * D * 2);
    __bf16* isf    = (__bf16*)alloc((size_t)NI * D * 2);
    __bf16* xtr    = (__bf16*)alloc((size_t)NT * D * 2);
    __bf16* xir    = (__bf16*)alloc((size_t)NI * D * 2);
    __bf16* ebT_t  = (__bf16*)alloc((size_t)B * K * St * 2);
    __bf16* ebT_i  = (__bf16*)alloc((size_t)B * K * Si * 2);
    float* AS_t  = (float*)alloc((size_t)NT * E * 4);
    float* AS_i  = (float*)alloc((size_t)NI * E * 4);
    float* BT_t  = (float*)alloc((size_t)NK * E * 4);
    float* BT_i  = (float*)alloc((size_t)NK * E * 4);
    float* tt_f  = (float*)alloc((size_t)NK * E * 4);
    float* ti_f  = (float*)alloc((size_t)NK * E * 4);
    float* stacc = (float*)alloc((size_t)B * K * D * 4);
    float* vlacc = (float*)alloc((size_t)B * K * D * 4);

    auto cvt = [&](const float* src, __bf16* dst, long n) {
        cvt_bf16_kernel<<<dim3((unsigned)((n + 255) / 256)), dim3(256), 0, stream>>>(src, dst, n);
    };

    // weight / value bf16 conversion
    cvt(t_su_w, wsu_t, (long)D * E);
    cvt(t_tp_w, wtp_t, (long)D * E);
    cvt(i_su_w, wsu_i, (long)D * E);
    cvt(i_tp_w, wtp_i, (long)D * E);
    cvt(t_m1_w, m1t_bf, (long)3 * E * E);
    cvt(i_m1_w, m1i_bf, (long)3 * E * E);
    cvt(tsp_w, wtsp, (long)D * D);
    cvt(isp_w, wisp, (long)D * D);
    cvt(text, xtr, (long)NT * D);
    cvt(image, xir, (long)NI * D);

    // layernorms -> bf16
    ln_kernel<true><<<NT, 256, 0, stream>>>(text,  t_sn_g, t_sn_b, xt_sn,  D);
    ln_kernel<true><<<NT, 256, 0, stream>>>(text,  tsn_g,  tsn_b,  xt_tsn, D);
    ln_kernel<true><<<NI, 256, 0, stream>>>(image, i_sn_g, i_sn_b, xi_sn,  D);
    ln_kernel<true><<<NI, 256, 0, stream>>>(image, isn_g,  isn_b,  xi_isn, D);
    ln_kernel<true><<<NK, 256, 0, stream>>>(lval0, t_tn_g, t_tn_b, lt_t,   D);
    ln_kernel<true><<<NK, 256, 0, stream>>>(lval0, i_tn_g, i_tn_b, lt_i,   D);

    // projections: s = LN(x)@su_w + su_b (bf16 out)
    gemm_bf16_kernel<true, false, true, false><<<dim3(E / 64, NT / 64, 1), 256, 0, stream>>>(
        xt_sn, wsu_t, t_su_b, nullptr, s_t, E, D, 0, 0, 0);
    gemm_bf16_kernel<true, false, true, false><<<dim3(E / 64, NI / 64, 1), 256, 0, stream>>>(
        xi_sn, wsu_i, i_su_b, nullptr, s_i, E, D, 0, 0, 0);
    // t = LN(lat)@tp_w + tp_b (f32 + bf16 out)
    gemm_bf16_kernel<true, true, true, false><<<dim3(E / 64, NK / 64, 1), 256, 0, stream>>>(
        lt_t, wtp_t, t_tp_b, tt_f, tt_bf, E, D, 0, 0, 0);
    gemm_bf16_kernel<true, true, true, false><<<dim3(E / 64, NK / 64, 1), 256, 0, stream>>>(
        lt_i, wtp_i, i_tp_b, ti_f, ti_bf, E, D, 0, 0, 0);
    // AS = s @ W1a + m1_b (f32 out)
    gemm_bf16_kernel<true, true, false, false><<<dim3(E / 64, NT / 64, 1), 256, 0, stream>>>(
        s_t, m1t_bf, t_m1_b, AS_t, nullptr, E, E, 0, 0, 0);
    gemm_bf16_kernel<true, true, false, false><<<dim3(E / 64, NI / 64, 1), 256, 0, stream>>>(
        s_i, m1i_bf, i_m1_b, AS_i, nullptr, E, E, 0, 0, 0);
    // BT = t @ W1b (f32 out)
    gemm_bf16_kernel<false, true, false, false><<<dim3(E / 64, NK / 64, 1), 256, 0, stream>>>(
        tt_bf, m1t_bf + (size_t)E * E, nullptr, BT_t, nullptr, E, E, 0, 0, 0);
    gemm_bf16_kernel<false, true, false, false><<<dim3(E / 64, NK / 64, 1), 256, 0, stream>>>(
        ti_bf, m1i_bf + (size_t)E * E, nullptr, BT_i, nullptr, E, E, 0, 0, 0);
    // feature projections (bf16 out)
    gemm_bf16_kernel<true, false, true, false><<<dim3(D / 64, NT / 64, 1), 256, 0, stream>>>(
        xt_tsn, wtsp, tsp_b, nullptr, tsf, D, D, 0, 0, 0);
    gemm_bf16_kernel<true, false, true, false><<<dim3(D / 64, NI / 64, 1), 256, 0, stream>>>(
        xi_isn, wisp, isp_b, nullptr, isf, D, D, 0, 0, 0);

    // fused edge scores
    edge_kernel<<<dim3(St / 32, K, B), 256, 0, stream>>>(
        s_t, AS_t, BT_t, tt_f, t_m1_w + (size_t)2 * E * E, t_m2_w, t_m2_b,
        tmask, out_t2l, ebT_t, St, K);
    edge_kernel<<<dim3(Si / 32, K, B), 256, 0, stream>>>(
        s_i, AS_i, BT_i, ti_f, i_m1_w + (size_t)2 * E * E, i_m2_w, i_m2_b,
        imask, out_i2l, ebT_i, Si, K);

    // residual accumulators
    hipMemcpyAsync(stacc, lstat0, (size_t)B * K * D * 4, hipMemcpyDeviceToDevice, stream);
    hipMemcpyAsync(vlacc, lval0,  (size_t)B * K * D * 4, hipMemcpyDeviceToDevice, stream);

    // e^T @ features (batched over B, accumulate into residual)
    gemm_bf16_kernel<false, true, false, true><<<dim3(D / 64, 1, B), 256, 0, stream>>>(
        ebT_t, tsf, nullptr, stacc, nullptr, D, St, (long)K * St, (long)St * D, (long)K * D);
    gemm_bf16_kernel<false, true, false, true><<<dim3(D / 64, 1, B), 256, 0, stream>>>(
        ebT_i, isf, nullptr, stacc, nullptr, D, Si, (long)K * Si, (long)Si * D, (long)K * D);
    gemm_bf16_kernel<false, true, false, true><<<dim3(D / 64, 1, B), 256, 0, stream>>>(
        ebT_t, xtr, nullptr, vlacc, nullptr, D, St, (long)K * St, (long)St * D, (long)K * D);
    gemm_bf16_kernel<false, true, false, true><<<dim3(D / 64, 1, B), 256, 0, stream>>>(
        ebT_i, xir, nullptr, vlacc, nullptr, D, Si, (long)K * Si, (long)Si * D, (long)K * D);

    // final layernorms -> outputs
    ln_kernel<false><<<NK, 256, 0, stream>>>(stacc, stn_g, stn_b, out_state, D);
    ln_kernel<false><<<NK, 256, 0, stream>>>(vlacc, vn_g,  vn_b,  out_value, D);
}
